// Attention_17935783428543
// MI455X (gfx1250) — matrix operand order repaired and checked by reading, not compiled, not run on hardware
//
#include <hip/hip_runtime.h>

// ---------------- problem constants ----------------
#define NB     4
#define CIN    128
#define HEADS  4
#define DHEAD  32
#define HID    128          // HEADS*DHEAD
#define O3     384          // 3*HID
#define NPIX   4096         // 64*64
#define SCALE_Q 0.1767766952966369f   // 32^-0.5

typedef __attribute__((ext_vector_type(16))) _Float16 v16h;
typedef __attribute__((ext_vector_type(8)))  _Float16 v8h;
typedef __attribute__((ext_vector_type(8)))  float    v8f;
typedef __attribute__((ext_vector_type(4)))  unsigned u32x4;
typedef __attribute__((ext_vector_type(8)))  unsigned u32x8;

__device__ __forceinline__ v8f zero8() {
  v8f z;
#pragma unroll
  for (int i = 0; i < 8; ++i) z[i] = 0.0f;
  return z;
}

// D = A*B + C, 16x16x32 f16 -> f32 accum (probe-confirmed builtin)
__device__ __forceinline__ v8f wmma16(v16h a, v16h b, v8f c) {
  return __builtin_amdgcn_wmma_f32_16x16x32_f16(
      /*neg_a=*/false, a, /*neg_b=*/false, b,
      /*c_mod=*/(short)0, c, /*reuse_a=*/false, /*reuse_b=*/false);
}

// A-matrix fragment loader (16x32 f16, row-major source, row stride 32 halfs):
// lane l -> row (l&15); elems 0..15 = K[16*(l>>4)..+15]
__device__ __forceinline__ v16h load_frag_a(const _Float16* row, int half) {
  const v8h lo = *(const v8h*)(row + 16 * half);
  const v8h hi = *(const v8h*)(row + 16 * half + 8);
  v16h r;
#pragma unroll
  for (int i = 0; i < 8; ++i) { r[i] = lo[i]; r[i + 8] = hi[i]; }
  return r;
}

// ---------------- Tensor Data Mover (CDNA5 ISA Sec.8) ----------------
// D# group 0: [1:0]=count(1), [63:32]=lds_addr, [120:64]=global_addr, [127:126]=type(2)
__device__ __forceinline__ u32x4 tdm_group0(unsigned lds_byte_off, unsigned long long gaddr) {
  u32x4 g;
  g[0] = 1u;                                        // count = 1 valid descriptor
  g[1] = lds_byte_off;                              // lds_addr (bytes)
  g[2] = (unsigned)(gaddr & 0xffffffffu);           // global_addr[31:0]
  g[3] = (unsigned)((gaddr >> 32) & 0x01ffffffu)    // global_addr[56:32]
       | (2u << 30);                                // type = 2 ("image")
  return g;
}
// D# group 1 for a 2D tile of 16-bit elements (data_size=1 -> 2 bytes):
//  [17:16]=data_size, [79:48]=tensor_dim0, [111:80]=tensor_dim1,
//  [127:112]=tile_dim0, [143:128]=tile_dim1, [207:160]=tensor_dim0_stride
__device__ __forceinline__ u32x8 tdm_group1_2d(unsigned td0, unsigned td1,
                                               unsigned tile0, unsigned tile1,
                                               unsigned long long stride0) {
  u32x8 g;
  g[0] = (1u << 16);                                        // data_size=2B, wg_mask=0
  g[1] = (td0 & 0xffffu) << 16;                             // tensor_dim0[15:0]
  g[2] = ((td0 >> 16) & 0xffffu) | ((td1 & 0xffffu) << 16); // td0[31:16] | td1[15:0]
  g[3] = ((td1 >> 16) & 0xffffu) | ((tile0 & 0xffffu) << 16);
  g[4] = (tile1 & 0xffffu);                                 // tile_dim1, tile_dim2=0
  g[5] = (unsigned)(stride0 & 0xffffffffu);                 // dim0 stride[31:0]
  g[6] = (unsigned)((stride0 >> 32) & 0xffffu);             // dim0 stride[47:32]
  g[7] = 0u;
  return g;
}
__device__ __forceinline__ void tdm_load(u32x4 g0, u32x8 g1) {
  asm volatile("tensor_load_to_lds %0, %1" :: "s"(g0), "s"(g1) : "memory");
}

// ============================================================
// Kernel 1: QKV projection.
//  qkv[o, i] = sum_c w_qkv[o,c] * x[b,c,i]
//  q -> [bh][i][d] f16 (pre-scaled), k -> [bh][i][d] f16, v -> [bh][d][i] f16 (transposed)
// ============================================================
__global__ __launch_bounds__(128) void qkv_kernel(
    const float* __restrict__ x, const float* __restrict__ wqkv,
    _Float16* __restrict__ qb, _Float16* __restrict__ kb, _Float16* __restrict__ vtb)
{
  __shared__ __align__(64) _Float16 As[64][32];   // weights [o][c-chunk]
  __shared__ __align__(64) _Float16 Bs[64][32];   // activations transposed [n][c-chunk]

  const int t = threadIdx.x;
  const int lane = t & 31, wave = t >> 5;
  const int nbase = blockIdx.x * 64;
  const int obase = blockIdx.y * 64;
  const int b     = blockIdx.z;

  v8f acc[4];
#pragma unroll
  for (int i = 0; i < 4; ++i) acc[i] = zero8();

  for (int kc = 0; kc < CIN; kc += 32) {
    {
      const int row = t >> 1, half = t & 1;
      const float* src = wqkv + (size_t)(obase + row) * CIN + kc + half * 16;
#pragma unroll
      for (int j = 0; j < 16; ++j) As[row][half * 16 + j] = (_Float16)src[j];
    }
    {
      const int c = t >> 2, q = t & 3;
      const float* src = x + ((size_t)b * CIN + kc + c) * NPIX + nbase + q * 16;
#pragma unroll
      for (int j = 0; j < 16; ++j) Bs[q * 16 + j][c] = (_Float16)src[j];
    }
    __syncthreads();

    const v16h a = load_frag_a(&As[wave * 16 + (lane & 15)][0], lane >> 4);
#pragma unroll
    for (int nt = 0; nt < 4; ++nt) {
      const v16h bfrag = *(const v16h*)(&Bs[nt * 16 + (lane & 15)][(lane >> 4) * 16]);
      acc[nt] = wmma16(a, bfrag, acc[nt]);
    }
    __syncthreads();
  }

#pragma unroll
  for (int nt = 0; nt < 4; ++nt) {
#pragma unroll
    for (int r = 0; r < 8; ++r) {
      const int o = obase + wave * 16 + r + 8 * (lane >> 4);
      const int i = nbase + nt * 16 + (lane & 15);
      const float v = acc[nt][r];
      const int sec = o >> 7;          // 0=q 1=k 2=v
      const int oo  = o & 127;
      const int h   = oo >> 5, d = oo & 31;
      const int bh  = b * HEADS + h;
      if (sec == 0)      qb[((size_t)bh * NPIX + i) * DHEAD + d] = (_Float16)(v * SCALE_Q);
      else if (sec == 1) kb[((size_t)bh * NPIX + i) * DHEAD + d] = (_Float16)v;
      else               vtb[((size_t)bh * DHEAD + d) * NPIX + i] = (_Float16)v;
    }
  }
}

// ============================================================
// Kernel 2: flash attention, TDM double-buffered K/V tiles in LDS.
//  Per block (4 waves, 64 queries): wave0 programs the TDM to DMA the shared
//  32-key K tile (32x32 f16) and V tile (32d x 32key, from v^T) into LDS;
//  TENSORcnt + barriers give a race-free 2-deep pipeline. All waves read
//  WMMA fragments from LDS (ds_load_b128), cutting L2 traffic 4x.
// ============================================================
__global__ __launch_bounds__(128) void attn_kernel(
    const _Float16* __restrict__ qb, const _Float16* __restrict__ kb,
    const _Float16* __restrict__ vtb, _Float16* __restrict__ yb)
{
  __shared__ __align__(64) _Float16 Kt[2][32][32];   // [buf][key][d]
  __shared__ __align__(64) _Float16 Vt[2][32][32];   // [buf][d][key]
  __shared__ __align__(64) _Float16 Ps[4][16][32];   // per-wave P tile

  const int t = threadIdx.x;
  const int lane = t & 31, wave = t >> 5;
  const int lh = lane & 15, half = lane >> 4;
  const int bh = blockIdx.y;                    // 0..15
  const int qtile = blockIdx.x * 64 + wave * 16;

  const _Float16* q  = qb  + (size_t)bh * NPIX * DHEAD;
  const _Float16* k  = kb  + (size_t)bh * NPIX * DHEAD;
  const _Float16* vt = vtb + (size_t)bh * DHEAD * NPIX;

  const v16h aq = load_frag_a(q + (size_t)(qtile + lh) * DHEAD, half);

  float m[8], ls[8];
#pragma unroll
  for (int r = 0; r < 8; ++r) { m[r] = -INFINITY; ls[r] = 0.0f; }
  v8f acc0 = zero8(), acc1 = zero8();

  // prologue: DMA tile 0 into buffer 0
  if (wave == 0) {
    tdm_load(tdm_group0((unsigned)(size_t)&Kt[0][0][0], (unsigned long long)(size_t)k),
             tdm_group1_2d(DHEAD, 32, DHEAD, 32, DHEAD));
    tdm_load(tdm_group0((unsigned)(size_t)&Vt[0][0][0], (unsigned long long)(size_t)vt),
             tdm_group1_2d(32, DHEAD, 32, DHEAD, NPIX));
  }

  for (int it = 0; it < NPIX / 32; ++it) {
    const int buf = it & 1;
    if (wave == 0) __builtin_amdgcn_s_wait_tensorcnt(0);
    __syncthreads();                       // tile `it` visible; buffer it-1 free

    if (wave == 0 && it + 1 < NPIX / 32) { // DMA tile it+1 into the other buffer
      const int k1 = (it + 1) * 32;
      tdm_load(tdm_group0((unsigned)(size_t)&Kt[buf ^ 1][0][0],
                          (unsigned long long)(size_t)(k + (size_t)k1 * DHEAD)),
               tdm_group1_2d(DHEAD, 32, DHEAD, 32, DHEAD));
      tdm_load(tdm_group0((unsigned)(size_t)&Vt[buf ^ 1][0][0],
                          (unsigned long long)(size_t)(vt + k1)),
               tdm_group1_2d(32, DHEAD, 32, DHEAD, NPIX));
    }

    // S = Q * K^T from the LDS tile
    const v16h bk0 = *(const v16h*)(&Kt[buf][lh][16 * half]);
    const v16h bk1 = *(const v16h*)(&Kt[buf][16 + lh][16 * half]);
    v8f s0 = wmma16(aq, bk0, zero8());
    v8f s1 = wmma16(aq, bk1, zero8());

    float p0[8], p1[8];
#pragma unroll
    for (int r = 0; r < 8; ++r) {
      float mx = fmaxf(s0[r], s1[r]);
      mx = fmaxf(mx, __shfl_xor(mx, 1, 32));
      mx = fmaxf(mx, __shfl_xor(mx, 2, 32));
      mx = fmaxf(mx, __shfl_xor(mx, 4, 32));
      mx = fmaxf(mx, __shfl_xor(mx, 8, 32));
      const float mn = fmaxf(m[r], mx);
      const float sc = __expf(m[r] - mn);
      const float e0 = __expf(s0[r] - mn);
      const float e1 = __expf(s1[r] - mn);
      float ps = e0 + e1;
      ps += __shfl_xor(ps, 1, 32);
      ps += __shfl_xor(ps, 2, 32);
      ps += __shfl_xor(ps, 4, 32);
      ps += __shfl_xor(ps, 8, 32);
      ls[r] = ls[r] * sc + ps;
      acc0[r] *= sc;
      acc1[r] *= sc;
      m[r] = mn;
      p0[r] = e0; p1[r] = e1;
    }

    // transpose P (C-layout -> A-layout) through per-wave LDS (same-wave DS is in-order)
#pragma unroll
    for (int r = 0; r < 8; ++r) {
      const int mm = r + 8 * half;
      Ps[wave][mm][lh]      = (_Float16)p0[r];
      Ps[wave][mm][16 + lh] = (_Float16)p1[r];
    }
    const v16h ap = load_frag_a(&Ps[wave][lh][0], half);

    const v16h bv0 = *(const v16h*)(&Vt[buf][lh][16 * half]);
    const v16h bv1 = *(const v16h*)(&Vt[buf][16 + lh][16 * half]);
    acc0 = wmma16(ap, bv0, acc0);
    acc1 = wmma16(ap, bv1, acc1);

    __syncthreads();                       // all waves done with buf before refill
  }

  // finalize: divide by row sums, pack 8 consecutive pixels per lane, store f16
  v8h o0, o1;
#pragma unroll
  for (int r = 0; r < 8; ++r) {
    const float inv = 1.0f / ls[r];
    o0[r] = (_Float16)(acc0[r] * inv);
    o1[r] = (_Float16)(acc1[r] * inv);
  }
  const int i0 = qtile + 8 * half;
  *(v8h*)(yb + ((size_t)bh * DHEAD + lh)      * NPIX + i0) = o0;
  *(v8h*)(yb + ((size_t)bh * DHEAD + 16 + lh) * NPIX + i0) = o1;
}

// ============================================================
// Kernel 3: output projection + bias.
//  out[b,c,i] = sum_o w_out[c,o] * y[b,o,i] + b_out[c]
// ============================================================
__global__ __launch_bounds__(128) void outproj_kernel(
    const _Float16* __restrict__ yb, const float* __restrict__ wout,
    const float* __restrict__ bout, float* __restrict__ out)
{
  __shared__ __align__(64) _Float16 As[64][32];   // w_out tile [c][o-chunk]
  __shared__ __align__(64) _Float16 Bs[64][32];   // y transposed [n][o-chunk]

  const int t = threadIdx.x;
  const int lane = t & 31, wave = t >> 5;
  const int nbase = blockIdx.x * 64;
  const int cbase = blockIdx.y * 64;
  const int b     = blockIdx.z;

  v8f acc[4];
#pragma unroll
  for (int i = 0; i < 4; ++i) acc[i] = zero8();

  for (int ko = 0; ko < HID; ko += 32) {
    {
      const int row = t >> 1, half = t & 1;
      const float* src = wout + (size_t)(cbase + row) * HID + ko + half * 16;
#pragma unroll
      for (int j = 0; j < 16; ++j) As[row][half * 16 + j] = (_Float16)src[j];
    }
    {
      const int o = t >> 2, qq = t & 3;
      const _Float16* src = yb + ((size_t)b * HID + ko + o) * NPIX + nbase + qq * 16;
#pragma unroll
      for (int j = 0; j < 16; ++j) Bs[qq * 16 + j][o] = src[j];
    }
    __syncthreads();

    const v16h a = load_frag_a(&As[wave * 16 + (lane & 15)][0], lane >> 4);
#pragma unroll
    for (int nt = 0; nt < 4; ++nt) {
      const v16h bfrag = *(const v16h*)(&Bs[nt * 16 + (lane & 15)][(lane >> 4) * 16]);
      acc[nt] = wmma16(a, bfrag, acc[nt]);
    }
    __syncthreads();
  }

#pragma unroll
  for (int nt = 0; nt < 4; ++nt) {
#pragma unroll
    for (int r = 0; r < 8; ++r) {
      const int c = cbase + wave * 16 + r + 8 * (lane >> 4);
      const int i = nbase + nt * 16 + (lane & 15);
      out[((size_t)b * CIN + c) * NPIX + i] = acc[nt][r] + bout[c];
    }
  }
}

// ============================================================
extern "C" void kernel_launch(void* const* d_in, const int* in_sizes, int n_in,
                              void* d_out, int out_size, void* d_ws, size_t ws_size,
                              hipStream_t stream) {
  const float* x     = (const float*)d_in[0];   // [4,128,64,64]
  const float* wqkv  = (const float*)d_in[1];   // [384,128]
  const float* wout  = (const float*)d_in[2];   // [128,128]
  const float* bout  = (const float*)d_in[3];   // [128]
  float* out = (float*)d_out;                   // [4,128,64,64]

  const size_t QELEMS = (size_t)NB * HEADS * NPIX * DHEAD;   // 2M halfs = 4MB
  _Float16* qb  = (_Float16*)d_ws;
  _Float16* kb  = qb  + QELEMS;
  _Float16* vtb = kb  + QELEMS;
  _Float16* yb  = vtb + QELEMS;                              // [4][128][4096] -> 4MB
  (void)ws_size; (void)in_sizes; (void)n_in; (void)out_size;

  qkv_kernel<<<dim3(NPIX / 64, O3 / 64, NB), 128, 0, stream>>>(x, wqkv, qb, kb, vtb);
  attn_kernel<<<dim3(NPIX / 64, NB * HEADS), 128, 0, stream>>>(qb, kb, vtb, yb);
  outproj_kernel<<<dim3(NPIX / 64, CIN / 64, NB), 128, 0, stream>>>(yb, wout, bout, out);
}
